// Encoder_48713519072053
// MI455X (gfx1250) — matrix operand order repaired and checked by reading, not compiled, not run on hardware
//
#include <hip/hip_runtime.h>
#include <hip/hip_bf16.h>

typedef __attribute__((ext_vector_type(16))) _Float16 v16h;
typedef __attribute__((ext_vector_type(8)))  float    v8f;

namespace {
constexpr int N_NODES = 16384;
constexpr int HID     = 256;
constexpr int MI      = 128;            // projected dim = WMMA K total
constexpr int NPOS    = 4;
// Stored embeddings are pre-scaled by sqrt(2 * log2(e)) so that
// <a,b> == log2(e) * sim / TEMP, letting us use raw v_exp_f32 (base-2).
constexpr float PRESCALE = 1.69864368f;   // sqrt(2 * 1.4426950408889634)
constexpr int TILE  = 16;
constexpr int MT    = 2;                       // A row-tiles per wave (32 rows)
constexpr int NTILES  = N_NODES / TILE;        // 1024 column tiles
constexpr int RTILES  = N_NODES / (TILE * MT); // 512 row tiles (32 rows each)
constexpr int WAVES_PER_BLK = 8;               // 256 threads, wave32
constexpr int CSPLIT = 8;                      // split column range for occupancy
constexpr int TILES_PER_SPLIT = NTILES / CSPLIT;

union FragH  { v16h v; uint4 q[2]; };          // 16 halves = 32B
union Pack4h { _Float16 h[4]; uint2 u; };      // 8B
union Pack8h { _Float16 h[8]; uint4 q; };      // 16B
} // namespace

// ---------------------------------------------------------------------------
// Zero the accumulators the other kernels atomically add into.
__global__ void Enc_init_kernel(float* __restrict__ rowsum1, float* __restrict__ rowsum2,
                                float* __restrict__ possum1, float* __restrict__ possum2,
                                float* __restrict__ out) {
  int i = blockIdx.x * blockDim.x + threadIdx.x;
  if (i < N_NODES) {
    rowsum1[i] = 0.f; rowsum2[i] = 0.f;
    possum1[i] = 0.f; possum2[i] = 0.f;
  }
  if (i == 0) out[0] = 0.f;
}

// ---------------------------------------------------------------------------
// p = PRESCALE * l2norm(relu(h @ W + b)), stored as f16 [N x 128].
// One wave per row; each lane owns 4 contiguous output columns.
__global__ void Enc_proj_kernel(const float* __restrict__ h1, const float* __restrict__ h2,
                                const float* __restrict__ W,  const float* __restrict__ b,
                                _Float16* __restrict__ e1h, _Float16* __restrict__ e2h) {
  const int lane = threadIdx.x & 31;
  const int wave = threadIdx.x >> 5;
  const int row  = blockIdx.x * WAVES_PER_BLK + wave;
  const float* __restrict__ h  = blockIdx.y ? h2 : h1;
  _Float16* __restrict__ out   = blockIdx.y ? e2h : e1h;
  const int col = lane * 4;

  float4 acc = *(const float4*)(b + col);
  const float* __restrict__ hrow = h + (size_t)row * HID;
#pragma unroll 4
  for (int k = 0; k < HID; k += 4) {
    float4 h4 = *(const float4*)(hrow + k);
    float4 w0 = *(const float4*)(W + (size_t)(k + 0) * MI + col);
    float4 w1 = *(const float4*)(W + (size_t)(k + 1) * MI + col);
    float4 w2 = *(const float4*)(W + (size_t)(k + 2) * MI + col);
    float4 w3 = *(const float4*)(W + (size_t)(k + 3) * MI + col);
    acc.x += h4.x * w0.x + h4.y * w1.x + h4.z * w2.x + h4.w * w3.x;
    acc.y += h4.x * w0.y + h4.y * w1.y + h4.z * w2.y + h4.w * w3.y;
    acc.z += h4.x * w0.z + h4.y * w1.z + h4.z * w2.z + h4.w * w3.z;
    acc.w += h4.x * w0.w + h4.y * w1.w + h4.z * w2.w + h4.w * w3.w;
  }
  acc.x = fmaxf(acc.x, 0.f); acc.y = fmaxf(acc.y, 0.f);
  acc.z = fmaxf(acc.z, 0.f); acc.w = fmaxf(acc.w, 0.f);

  float ss = acc.x * acc.x + acc.y * acc.y + acc.z * acc.z + acc.w * acc.w;
  ss += __shfl_xor(ss, 16, 32);
  ss += __shfl_xor(ss,  8, 32);
  ss += __shfl_xor(ss,  4, 32);
  ss += __shfl_xor(ss,  2, 32);
  ss += __shfl_xor(ss,  1, 32);
  const float rn = rsqrtf(ss) * PRESCALE;

  Pack4h p;
  p.h[0] = (_Float16)(acc.x * rn);
  p.h[1] = (_Float16)(acc.y * rn);
  p.h[2] = (_Float16)(acc.z * rn);
  p.h[3] = (_Float16)(acc.w * rn);
  *(uint2*)(out + (size_t)row * MI + col) = p.u;
}

// ---------------------------------------------------------------------------
// rowsum[i] = sum_j exp2(<A_i, B_j>) = sum_j exp(sim_ij / TEMP)
// (operands carry the sqrt(2*log2e) pre-scale).
// gridDim.y selects direction (A,B) = (e1,e2) or (e2,e1) since sim2 = sim^T.
// Each wave owns TWO 16-row tiles (32 rows): every loaded B fragment feeds two
// independent v_wmma_f32_16x16x32_f16 chains -> 8 WMMA per 8 b128 loads.
// "#pragma unroll 1" on the column loop keeps live state under the VGPR
// budget (no scratch spills); ILP comes from the 8 loads + 2 WMMA chains.
__global__ void Enc_rowsum_kernel(const _Float16* __restrict__ e1h,
                                  const _Float16* __restrict__ e2h,
                                  float* __restrict__ rowsum1,
                                  float* __restrict__ rowsum2) {
  const int lane  = threadIdx.x & 31;
  const int wave  = threadIdx.x >> 5;
  const int split = blockIdx.x & (CSPLIT - 1);
  const int rtile = (blockIdx.x / CSPLIT) * WAVES_PER_BLK + wave;
  const int dir   = blockIdx.y;
  const _Float16* __restrict__ A = dir ? e2h : e1h;
  const _Float16* __restrict__ B = dir ? e1h : e2h;
  float* __restrict__ out        = dir ? rowsum2 : rowsum1;

  const int m    = lane & 15;   // A: row within tile | B: col within tile
  const int hi   = lane >> 4;   // lane-half selects K sub-range per ISA layout
  const int row0 = rtile * (TILE * MT);

  // A fragments: lanes 0-15 hold K {0..7, 16..23}, lanes 16-31 hold {8..15, 24..31}
  FragH a0[4], a1[4];
  const _Float16* __restrict__ pa0 = A + (size_t)(row0 + m) * MI + hi * 8;
  const _Float16* __restrict__ pa1 = pa0 + (size_t)TILE * MI;
#pragma unroll
  for (int kc = 0; kc < 4; ++kc) {
    a0[kc].q[0] = *(const uint4*)(pa0 + kc * 32);
    a0[kc].q[1] = *(const uint4*)(pa0 + kc * 32 + 16);
    a1[kc].q[0] = *(const uint4*)(pa1 + kc * 32);
    a1[kc].q[1] = *(const uint4*)(pa1 + kc * 32 + 16);
  }

  float rs0[8] = {0.f, 0.f, 0.f, 0.f, 0.f, 0.f, 0.f, 0.f};
  float rs1[8] = {0.f, 0.f, 0.f, 0.f, 0.f, 0.f, 0.f, 0.f};
  const int t0 = split * TILES_PER_SPLIT;
#pragma unroll 1
  for (int t = t0; t < t0 + TILES_PER_SPLIT; ++t) {
    // B fragment: lanes 0-15 hold K {0..7, 16..23}, lanes 16-31 hold {8..15, 24..31}
    const _Float16* __restrict__ pb = B + (size_t)(t * TILE + m) * MI + hi * 8;
    __builtin_prefetch(pb + 4 * TILE * MI, 0, 1);   // global_prefetch_b8 ahead
    FragH bf[4];
#pragma unroll
    for (int kc = 0; kc < 4; ++kc) {
      bf[kc].q[0] = *(const uint4*)(pb + kc * 32);
      bf[kc].q[1] = *(const uint4*)(pb + kc * 32 + 16);
    }
    v8f c0 = {};
    v8f c1 = {};
    // Interleave the two chains so each bf[kc] dies after its second use.
#pragma unroll
    for (int kc = 0; kc < 4; ++kc) {
      c0 = __builtin_amdgcn_wmma_f32_16x16x32_f16(false, a0[kc].v, false, bf[kc].v,
                                                  (short)0, c0, false, false);
      c1 = __builtin_amdgcn_wmma_f32_16x16x32_f16(false, a1[kc].v, false, bf[kc].v,
                                                  (short)0, c1, false, false);
    }
    // C layout: VGPR i holds M=i (lanes 0-15) / M=8+i (lanes 16-31), N = lane&15
#pragma unroll
    for (int i = 0; i < 8; ++i) {
      rs0[i] += __builtin_amdgcn_exp2f(c0[i]);   // bare v_exp_f32, no mul
      rs1[i] += __builtin_amdgcn_exp2f(c1[i]);
    }
  }

  // Reduce over the 16 N-lanes of each half, then one atomic per row per split.
#pragma unroll
  for (int i = 0; i < 8; ++i) {
    float v0 = rs0[i];
    v0 += __shfl_xor(v0, 1, 32);
    v0 += __shfl_xor(v0, 2, 32);
    v0 += __shfl_xor(v0, 4, 32);
    v0 += __shfl_xor(v0, 8, 32);
    float v1 = rs1[i];
    v1 += __shfl_xor(v1, 1, 32);
    v1 += __shfl_xor(v1, 2, 32);
    v1 += __shfl_xor(v1, 4, 32);
    v1 += __shfl_xor(v1, 8, 32);
    if (m == 0) {
      atomicAdd(&out[row0 + hi * 8 + i], v0);
      atomicAdd(&out[row0 + TILE + hi * 8 + i], v1);
    }
  }
}

// ---------------------------------------------------------------------------
// Positive pairs: spos_d[e] = exp2(dot) = exp(sim/TEMP), possum_d[row] += spos_d[e].
__global__ void Enc_pair_kernel(const _Float16* __restrict__ e1h,
                                const _Float16* __restrict__ e2h,
                                const int* __restrict__ pos_row,
                                const int* __restrict__ pos_col,
                                float* __restrict__ spos1, float* __restrict__ spos2,
                                float* __restrict__ possum1, float* __restrict__ possum2,
                                int E) {
  int e = blockIdx.x * blockDim.x + threadIdx.x;
  if (e >= E) return;
  const int r = pos_row[e], c = pos_col[e];
  const _Float16* __restrict__ a1 = e1h + (size_t)r * MI;
  const _Float16* __restrict__ b1 = e2h + (size_t)c * MI;
  const _Float16* __restrict__ a2 = e2h + (size_t)r * MI;
  const _Float16* __restrict__ b2 = e1h + (size_t)c * MI;
  float d1 = 0.f, d2 = 0.f;
#pragma unroll 4
  for (int k = 0; k < MI; k += 8) {
    Pack8h x1, y1, x2, y2;
    x1.q = *(const uint4*)(a1 + k); y1.q = *(const uint4*)(b1 + k);
    x2.q = *(const uint4*)(a2 + k); y2.q = *(const uint4*)(b2 + k);
#pragma unroll
    for (int j = 0; j < 8; ++j) {
      d1 += (float)x1.h[j] * (float)y1.h[j];
      d2 += (float)x2.h[j] * (float)y2.h[j];
    }
  }
  const float s1 = __builtin_amdgcn_exp2f(d1);  // dot already carries log2e/TEMP
  const float s2 = __builtin_amdgcn_exp2f(d2);
  spos1[e] = s1; spos2[e] = s2;
  atomicAdd(&possum1[r], s1);
  atomicAdd(&possum2[r], s2);
}

// ---------------------------------------------------------------------------
// loss = -(S1 + S2) / (8N), S_d = sum_e log(spos_d / (rowsum_d - possum_d)).
__global__ void Enc_loss_kernel(const int* __restrict__ pos_row,
                                const float* __restrict__ spos1, const float* __restrict__ spos2,
                                const float* __restrict__ rowsum1, const float* __restrict__ rowsum2,
                                const float* __restrict__ possum1, const float* __restrict__ possum2,
                                float* __restrict__ out, int E) {
  __shared__ float sm[256];
  int e = blockIdx.x * blockDim.x + threadIdx.x;
  float t = 0.f;
  if (e < E) {
    const int r = pos_row[e];
    t = __logf(spos1[e] / (rowsum1[r] - possum1[r]))
      + __logf(spos2[e] / (rowsum2[r] - possum2[r]));
  }
  sm[threadIdx.x] = t;
  __syncthreads();
  for (int s = 128; s > 0; s >>= 1) {
    if ((int)threadIdx.x < s) sm[threadIdx.x] += sm[threadIdx.x + s];
    __syncthreads();
  }
  if (threadIdx.x == 0)
    atomicAdd(out, sm[0] * (-1.0f / (2.0f * (float)NPOS * (float)N_NODES)));
}

// ---------------------------------------------------------------------------
extern "C" void kernel_launch(void* const* d_in, const int* in_sizes, int n_in,
                              void* d_out, int out_size, void* d_ws, size_t ws_size,
                              hipStream_t stream) {
  const float* h_v1 = (const float*)d_in[0];
  const float* h_v2 = (const float*)d_in[1];
  const float* W    = (const float*)d_in[2];
  const float* b    = (const float*)d_in[3];
  const int* pos_row = (const int*)d_in[4];
  const int* pos_col = (const int*)d_in[5];
  const int E = in_sizes[4];

  char* ws = (char*)d_ws;
  _Float16* e1h  = (_Float16*)ws;
  _Float16* e2h  = (_Float16*)(ws + (size_t)N_NODES * MI * 2);
  float* rowsum1 = (float*)(ws + (size_t)N_NODES * MI * 4);
  float* rowsum2 = rowsum1 + N_NODES;
  float* possum1 = rowsum2 + N_NODES;
  float* possum2 = possum1 + N_NODES;
  float* spos1   = possum2 + N_NODES;
  float* spos2   = spos1 + E;
  float* out = (float*)d_out;

  Enc_init_kernel<<<dim3((N_NODES + 255) / 256), dim3(256), 0, stream>>>(
      rowsum1, rowsum2, possum1, possum2, out);

  Enc_proj_kernel<<<dim3(N_NODES / WAVES_PER_BLK, 2), dim3(256), 0, stream>>>(
      h_v1, h_v2, W, b, e1h, e2h);

  Enc_rowsum_kernel<<<dim3((RTILES / WAVES_PER_BLK) * CSPLIT, 2), dim3(256), 0, stream>>>(
      e1h, e2h, rowsum1, rowsum2);

  Enc_pair_kernel<<<dim3((E + 255) / 256), dim3(256), 0, stream>>>(
      e1h, e2h, pos_row, pos_col, spos1, spos2, possum1, possum2, E);

  Enc_loss_kernel<<<dim3((E + 255) / 256), dim3(256), 0, stream>>>(
      pos_row, spos1, spos2, rowsum1, rowsum2, possum1, possum2, out, E);
}
